// NeRF_net_5463198400602
// MI455X (gfx1250) — matrix operand order repaired and checked by reading, not compiled, not run on hardware
//
#include <hip/hip_runtime.h>
#include <hip/hip_bf16.h>

typedef __attribute__((ext_vector_type(8)))  _Float16 v8h;
typedef __attribute__((ext_vector_type(16))) _Float16 v16h;
typedef __attribute__((ext_vector_type(8)))  float    v8f;

#define NSAMP   (4096 * 64)   // total samples
#define AST     296           // activation row stride (f16), 592B = 37*16B aligned
#define WST     40            // weight LDS row stride (f16), 80B = 5*16B aligned
#define MROWS   32            // samples per workgroup (2 row-tiles of 16)
#define BLK     64            // threads per block = 2 wave32 (waves split columns)
#define WBUF    (272 * WST)   // one weight buffer in f16 elements
#define WS_F16  719360        // total f16 elements of converted weights in d_ws

union FragU { v16h v; v8h h[2]; };

struct P {
    const float* x;
    const float* d;
    const float* W[11];   // W0..W8, Wc0, Wc1
    const float* b[11];   // b0..b8, bc0, bc1
    float*       out;     // [NSAMP] sigma, then [NSAMP*3] c
    _Float16*    ws;      // converted/pre-chunked f16 weights
    int          use_ws;
};

// ---------------------------------------------------------------------------
// Pre-pass: convert f32 weights -> f16 in d_ws, laid out per K-chunk of 32,
// column-major with row stride WST (ready for straight async LDS copy +
// aligned 16B B-fragment loads). Zero-pads K and N to WMMA-friendly shapes.
// ---------------------------------------------------------------------------
__global__ void __launch_bounds__(256) cvt_weights(P p) {
    const int CH[11] = {1,8,8,8,8,8,8,8,8,9,4};                      // K chunks
    const int KO[11] = {30,256,256,256,256,256,256,256,256,268,128}; // K orig
    const int NO[11] = {256,256,256,256,256,256,256,256,257,128,3};  // N orig
    const int NP[11] = {256,256,256,256,256,256,256,256,272,128,16}; // N padded
    const size_t OFF[11] = {0,10240,92160,174080,256000,337920,
                            419840,501760,583680,670720,716800};

    int layer = 0, rem = blockIdx.x;
    while (rem >= CH[layer]) { rem -= CH[layer]; ++layer; }

    const float* W  = p.W[layer];
    const int Ko = KO[layer], No = NO[layer], Np = NP[layer];
    _Float16* dst = p.ws + OFF[layer] + (size_t)rem * Np * WST;
    const int kbase = rem * 32;

    for (int i = threadIdx.x; i < Np * 32; i += 256) {
        const int n = i >> 5, k = i & 31, kg = kbase + k;
        float v = (kg < Ko && n < No) ? W[(size_t)kg * No + n] : 0.0f;
        dst[n * WST + k] = (_Float16)v;
    }
}

// ---------------------------------------------------------------------------
// Issue one weight K-chunk as an async global->LDS DMA (gfx1250, ASYNCcnt).
// Packet count per wave is made uniform (clamped index) so s_wait_asynccnt
// bookkeeping is exact for the odd-sized layers too.
// ---------------------------------------------------------------------------
template<int NT>
__device__ __forceinline__ void issue_wchunk(_Float16* dstBuf,
                                             const _Float16* wsBase, int kc) {
    constexpr int NPKT = NT * 16 * WST / 8;          // 16B packets per chunk
    constexpr int PPT  = (NPKT + BLK - 1) / BLK;     // uniform packets/thread
    const _Float16* gsrc = wsBase + (size_t)kc * NT * 16 * WST;
    const unsigned lds0 = (unsigned)(unsigned long long)(void*)dstBuf;
#pragma unroll
    for (int j = 0; j < PPT; ++j) {
        int i = (int)threadIdx.x + j * BLK;
        if (NPKT % BLK != 0 && i >= NPKT) i = NPKT - 1;  // benign duplicate
        unsigned loff = lds0 + (unsigned)i * 16u;
        unsigned goff = (unsigned)i * 16u;
        asm volatile("global_load_async_to_lds_b128 %0, %1, %2"
                     :: "v"(loff), "v"(goff), "s"(gsrc) : "memory");
    }
}

// ---------------------------------------------------------------------------
// One dense layer on a 32-sample tile; 2 waves split the column tiles,
// each wave computes all 32 rows (2 row-tiles) for its half of N.
// Weight chunks are double-buffered: DMA of chunk k+1 overlaps WMMAs of k.
//   NT  : number of 16-wide output column tiles (N = NT*16, padded)
//   NC  : number of K-chunks of 32
//   EPI : 0 = relu -> f16 actOut[col]
//         1 = layer8: col0 -> sigma (f32 global), col 1..256 -> actOut[col-1]
//         2 = head:   col<3 -> c (f32 global)
// ---------------------------------------------------------------------------
template<int NT, int NC, int EPI>
__device__ __forceinline__ void layer_gemm(
    const _Float16* __restrict__ actIn, _Float16* __restrict__ actOut,
    _Float16* s_wAll, float* s_bias,
    const float* __restrict__ Wf32, int Ko, int No,
    const float* __restrict__ bias,
    const _Float16* __restrict__ wsBase, int use_ws,
    float* __restrict__ out, int s0)
{
    const int tid  = threadIdx.x;
    const int lane = tid & 31;
    const int wave = tid >> 5;
    const int lm   = lane & 15;   // col-within-tile / row-within-tile
    const int hsel = lane >> 4;   // which K/M half this lane services
    constexpr int NTW = (NT + 1) / 2;   // column tiles per wave (ceil)

    __syncthreads();   // all waves done with previous layer's s_w/s_bias reads
    if (use_ws) issue_wchunk<NT>(s_wAll, wsBase, 0);   // prime the pipeline

    for (int i = tid; i < NT * 16; i += BLK)
        s_bias[i] = (i < No) ? bias[i] : 0.0f;

    v8f acc[2][NTW];
#pragma unroll
    for (int rt = 0; rt < 2; ++rt)
#pragma unroll
        for (int nt = 0; nt < NTW; ++nt)
#pragma unroll
            for (int j = 0; j < 8; ++j) acc[rt][nt][j] = 0.0f;

#pragma unroll 1
    for (int kc = 0; kc < NC; ++kc) {
        _Float16* wcur = s_wAll + (kc & 1) * WBUF;
        if (use_ws) {
            // my portion of chunk kc has landed in LDS
            asm volatile("s_wait_asynccnt 0x0" ::: "memory");
            __syncthreads();   // everyone's portion landed; buf^1 free to refill
            if (kc + 1 < NC)
                issue_wchunk<NT>(s_wAll + ((kc + 1) & 1) * WBUF, wsBase, kc + 1);
        } else {
            __syncthreads();
            // fallback: transpose+convert f32 weights on the fly
            for (int i = tid; i < NT * 16 * 32; i += BLK) {
                const int n = i >> 5, k = i & 31, kg = kc * 32 + k;
                float v = (kg < Ko && n < No) ? Wf32[(size_t)kg * No + n] : 0.0f;
                wcur[n * WST + k] = (_Float16)v;
            }
            if (kc + 1 < NC)
                __builtin_prefetch(Wf32 + (size_t)(kc + 1) * 32 * No, 0, 1);
            __syncthreads();
        }

        // A fragments (16x32 f16), one per row-tile: lane = M; two aligned
        // 16B runs at K = kc*32 + hsel*8 + {0..7} and +16
        FragU a[2];
#pragma unroll
        for (int rt = 0; rt < 2; ++rt) {
            const int ab = (rt * 16 + lm) * AST + kc * 32 + hsel * 8;
            a[rt].h[0] = *(const v8h*)(actIn + ab);
            a[rt].h[1] = *(const v8h*)(actIn + ab + 16);
        }

#pragma unroll
        for (int nt = 0; nt < NTW; ++nt) {
            const int g = wave * NTW + nt;          // global column tile
            if (NT % 2 == 1 && g >= NT) continue;   // wave-uniform skip (odd NT)
            // B fragment (32x16 f16): lane = N; K = hsel*8 + {0..7} and +16
            FragU bfr;
            const int wb = (g * 16 + lm) * WST + hsel * 8;
            bfr.h[0] = *(const v8h*)(wcur + wb);
            bfr.h[1] = *(const v8h*)(wcur + wb + 16);
#pragma unroll
            for (int rt = 0; rt < 2; ++rt)
                acc[rt][nt] = __builtin_amdgcn_wmma_f32_16x16x32_f16(
                    false, a[rt].v, false, bfr.v, (short)0, acc[rt][nt],
                    false, false);
        }
    }

    // epilogue: bias + ReLU; C layout: VGPR r -> row rt*16+r+hsel*8,
    // col = tile*16 + lm
#pragma unroll
    for (int nt = 0; nt < NTW; ++nt) {
        const int g = wave * NTW + nt;
        if (NT % 2 == 1 && g >= NT) continue;
        const int col = g * 16 + lm;
        const float bb = s_bias[col];
#pragma unroll
        for (int rt = 0; rt < 2; ++rt)
#pragma unroll
            for (int r = 0; r < 8; ++r) {
                const int row = rt * 16 + r + hsel * 8;
                float v = fmaxf(acc[rt][nt][r] + bb, 0.0f);
                if (EPI == 0) {
                    actOut[row * AST + col] = (_Float16)v;
                } else if (EPI == 1) {
                    if (col == 0)        out[s0 + row] = v;   // sigma
                    else if (col <= 256) actOut[row * AST + (col - 1)] = (_Float16)v;
                } else {
                    if (col < 3) out[NSAMP + (size_t)(s0 + row) * 3 + col] = v;
                }
            }
    }
}

// ---------------------------------------------------------------------------
__global__ void __launch_bounds__(BLK) nerf_fwd(P p) {
    __shared__ __align__(16) _Float16 sA[MROWS * AST];
    __shared__ __align__(16) _Float16 sB[MROWS * AST];
    __shared__ __align__(16) _Float16 s_w[2 * WBUF];   // double-buffered chunk
    __shared__ float s_bias[272];

    const int s0  = blockIdx.x * MROWS;
    const int tid = threadIdx.x;

    // stage x (30 features, zero-padded to 32) as f16 into sA
    for (int i = tid; i < MROWS * 32; i += BLK) {
        const int m = i >> 5, k = i & 31;
        float v = (k < 30) ? p.x[(size_t)(s0 + m) * 30 + k] : 0.0f;
        sA[m * AST + k] = (_Float16)v;
    }

    const size_t OFF[11] = {0,10240,92160,174080,256000,337920,
                            419840,501760,583680,670720,716800};

    // trunk: 30->256, 7x 256->256, 256->257(split sigma/fx)
    layer_gemm<16,1,0>(sA, sB, s_w, s_bias, p.W[0], 30, 256, p.b[0],
                       p.ws + OFF[0], p.use_ws, p.out, s0);
    layer_gemm<16,8,0>(sB, sA, s_w, s_bias, p.W[1], 256, 256, p.b[1],
                       p.ws + OFF[1], p.use_ws, p.out, s0);
    layer_gemm<16,8,0>(sA, sB, s_w, s_bias, p.W[2], 256, 256, p.b[2],
                       p.ws + OFF[2], p.use_ws, p.out, s0);
    layer_gemm<16,8,0>(sB, sA, s_w, s_bias, p.W[3], 256, 256, p.b[3],
                       p.ws + OFF[3], p.use_ws, p.out, s0);
    layer_gemm<16,8,0>(sA, sB, s_w, s_bias, p.W[4], 256, 256, p.b[4],
                       p.ws + OFF[4], p.use_ws, p.out, s0);
    layer_gemm<16,8,0>(sB, sA, s_w, s_bias, p.W[5], 256, 256, p.b[5],
                       p.ws + OFF[5], p.use_ws, p.out, s0);
    layer_gemm<16,8,0>(sA, sB, s_w, s_bias, p.W[6], 256, 256, p.b[6],
                       p.ws + OFF[6], p.use_ws, p.out, s0);
    layer_gemm<16,8,0>(sB, sA, s_w, s_bias, p.W[7], 256, 256, p.b[7],
                       p.ws + OFF[7], p.use_ws, p.out, s0);
    layer_gemm<17,8,1>(sA, sB, s_w, s_bias, p.W[8], 256, 257, p.b[8],
                       p.ws + OFF[8], p.use_ws, p.out, s0);

    // append d (12 f32 -> f16) at cols 256..267, zero-pad to 288
    for (int i = tid; i < MROWS * 32; i += BLK) {
        const int m = i >> 5, k = i & 31;
        float v = (k < 12) ? p.d[(size_t)(s0 + m) * 12 + k] : 0.0f;
        sB[m * AST + 256 + k] = (_Float16)v;
    }

    // head: 268(pad 288)->128, then 128->3(pad 16)
    layer_gemm<8,9,0>(sB, sA, s_w, s_bias, p.W[9], 268, 128, p.b[9],
                      p.ws + OFF[9], p.use_ws, p.out, s0);
    layer_gemm<1,4,2>(sA, sB, s_w, s_bias, p.W[10], 128, 3, p.b[10],
                      p.ws + OFF[10], p.use_ws, p.out, s0);
}

// ---------------------------------------------------------------------------
extern "C" void kernel_launch(void* const* d_in, const int* in_sizes, int n_in,
                              void* d_out, int out_size, void* d_ws, size_t ws_size,
                              hipStream_t stream) {
    P p;
    p.x = (const float*)d_in[0];
    p.d = (const float*)d_in[1];
    for (int i = 0; i < 9; ++i) {
        p.W[i] = (const float*)d_in[2 + 2 * i];
        p.b[i] = (const float*)d_in[3 + 2 * i];
    }
    p.W[9]  = (const float*)d_in[20];  p.b[9]  = (const float*)d_in[21];  // Wc0,bc0
    p.W[10] = (const float*)d_in[22];  p.b[10] = (const float*)d_in[23];  // Wc1,bc1
    p.out = (float*)d_out;
    p.ws  = (_Float16*)d_ws;
    p.use_ws = (ws_size >= (size_t)WS_F16 * sizeof(_Float16)) ? 1 : 0;

    if (p.use_ws)
        cvt_weights<<<78, 256, 0, stream>>>(p);
    nerf_fwd<<<NSAMP / MROWS, BLK, 0, stream>>>(p);
}
